// VolumeFeatureAggregator_85109071938164
// MI455X (gfx1250) — compile-verified, matrix operand order checked by reading, not run on hardware
//
#include <hip/hip_runtime.h>
#include <hip/hip_bf16.h>

typedef __attribute__((ext_vector_type(16))) _Float16 v16h;
typedef __attribute__((ext_vector_type(8)))  float    v8f;

#define NPTS    32768
#define BATCH   8
#define NGRID   32768
#define C_IN    131
#define C_HID   1024
#define C_OUT   128
#define KT1     5        // 160 / 32 k-tiles for GEMM1
#define NT1     64       // 1024 / 16 n-tiles for GEMM1
#define KT2     32       // 1024 / 32 k-tiles for GEMM2
#define NT2     8        // 128 / 16 n-tiles for GEMM2
#define MT      4        // 4 x 16 = 64 points per workgroup
#define MTILE   64
#define XSTRIDE 168      // 160 + 8 pad (halves) to break LDS bank conflicts
#define HSTRIDE 1032     // 1024 + 8 pad
#define BN_EPS  1e-5f
#define ENC_EMPTY 0x007FFFFFu   // encode(-inf): "no point hit this voxel"

// ---- order-preserving float <-> uint encoding (for atomic max on u32) ----
__device__ __forceinline__ unsigned enc_f32(float f) {
  unsigned u = __float_as_uint(f);
  return (u & 0x80000000u) ? ~u : (u | 0x80000000u);
}
__device__ __forceinline__ float dec_f32(unsigned u) {
  unsigned b = (u & 0x80000000u) ? (u & 0x7FFFFFFFu) : ~u;
  return __uint_as_float(b);
}

// K position of packed half 'e' (0..15) for lane 'l' (0..31) within one
// f16 WMMA 16x32 A-tile / 32x16 B-tile (CDNA5 ISA 7.12.2 layout).
__device__ __forceinline__ int kmap(int l, int e) {
  int v = e >> 1, h = e & 1;
  return ((v & 4) ? 16 : 0) + ((l >= 16) ? 8 : 0) + ((v & 3) << 1) + h;
}

// Gather one WMMA A-tile (16 rows x 32 K, f16) from row-major LDS.
// K pairs (2h,2h+1) are contiguous -> lowers to ds_load_b128 pairs.
__device__ __forceinline__ v16h lds_load_a(const _Float16* base, int rowStride,
                                           int m, int hi8, int kbase) {
  v16h a;
  unsigned* au = reinterpret_cast<unsigned*>(&a);
#pragma unroll
  for (int v = 0; v < 8; ++v) {
    int k = kbase + ((v & 4) ? 16 : 0) + hi8 + ((v & 3) << 1);
    au[v] = *reinterpret_cast<const unsigned*>(base + m * rowStride + k);
  }
  return a;
}

// ---------------- weight repack kernels (f32 -> f16, WMMA B lane layout) ----
__global__ void pack_w1_kernel(const float* __restrict__ W1, _Float16* __restrict__ w1p) {
  int s = blockIdx.x * blockDim.x + threadIdx.x;
  if (s >= KT1 * NT1 * 512) return;
  int kt  = s / (NT1 * 512);
  int rem = s % (NT1 * 512);
  int nt  = rem >> 9;
  int idx = rem & 511;
  int l = idx >> 4, e = idx & 15;
  int k = kt * 32 + kmap(l, e);
  int n = nt * 16 + (l & 15);
  float val = (k < C_IN) ? W1[k * C_HID + n] : 0.0f;   // zero-pad K 131..159
  w1p[s] = (_Float16)val;
}

__global__ void pack_w2_kernel(const float* __restrict__ W2, _Float16* __restrict__ w2p) {
  int s = blockIdx.x * blockDim.x + threadIdx.x;
  if (s >= KT2 * NT2 * 512) return;
  int kt  = s / (NT2 * 512);
  int rem = s % (NT2 * 512);
  int nt  = rem >> 9;
  int idx = rem & 511;
  int l = idx >> 4, e = idx & 15;
  int k = kt * 32 + kmap(l, e);
  int n = nt * 16 + (l & 15);
  w2p[s] = (_Float16)W2[k * C_OUT + n];
}

// ---------------- grid init / finalize ----------------
__global__ void init_out_kernel(unsigned* __restrict__ outU, int n) {
  int i = blockIdx.x * blockDim.x + threadIdx.x;
  if (i < n) outU[i] = ENC_EMPTY;
}

__global__ void finalize_out_kernel(unsigned* __restrict__ outU, float* __restrict__ outF, int n) {
  int i = blockIdx.x * blockDim.x + threadIdx.x;
  if (i < n) {
    unsigned u = outU[i];
    outF[i] = (u == ENC_EMPTY) ? 0.0f : dec_f32(u);
  }
}

// ---------------- fused MLP + scatter-max ----------------
// One workgroup = 8 waves = 64 points. Each weight B-tile register load is
// reused across 4 M-tiles, cutting L2 weight streaming 4x vs M=16.
__global__ __launch_bounds__(256)
void fused_mlp_scatter_kernel(
    const float* __restrict__ xyz, const float* __restrict__ pf,
    const float* __restrict__ b1,  const float* __restrict__ g1,
    const float* __restrict__ be1, const float* __restrict__ m1,
    const float* __restrict__ v1,
    const float* __restrict__ b2,  const float* __restrict__ g2,
    const float* __restrict__ be2, const float* __restrict__ m2,
    const float* __restrict__ v2,
    const _Float16* __restrict__ w1p, const _Float16* __restrict__ w2p,
    unsigned* __restrict__ outU) {
  __shared__ alignas(16) _Float16 Xs [MTILE * XSTRIDE];   // 64 pts x 160 ch
  __shared__ alignas(16) _Float16 H1s[MTILE * HSTRIDE];   // 64 pts x 1024 hidden
  __shared__ int voxs[MTILE];

  const int b    = blockIdx.x >> 9;          // 512 tiles of 64 points per batch
  const int p0   = (blockIdx.x & 511) << 6;  // first point of tile
  const int t    = threadIdx.x;
  const int lane = t & 31;
  const int wave = t >> 5;
  const int mrow = lane & 15;                // N column (C layout) / A row
  const int hi8  = (lane >> 4) << 3;         // +8 row/K shift for upper half-wave

  // ---- stage X tile: fully coalesced (64 consecutive pts per row) ----
  {
    const int m  = t & 63;       // point within tile
    const int kg = t >> 6;       // 0..3
    const int p  = p0 + m;
#pragma unroll
    for (int pass = 0; pass < 40; ++pass) {
      int k = pass * 4 + kg;     // 0..159
      float val = 0.0f;
      if (k < 3)         val = xyz[((size_t)b * 3 + k) * NPTS + p];
      else if (k < C_IN) val = pf[((size_t)b * C_OUT + (k - 3)) * NPTS + p];
      Xs[m * XSTRIDE + k] = (_Float16)val;
    }
    if (t < MTILE) {   // voxel index per point
      float x = xyz[((size_t)b * 3 + 0) * NPTS + p];
      float y = xyz[((size_t)b * 3 + 1) * NPTS + p];
      float z = xyz[((size_t)b * 3 + 2) * NPTS + p];
      int ix = min(max((int)floorf((x + 0.5f) * 32.0f), 0), 31);
      int iy = min(max((int)floorf((y + 0.5f) * 32.0f), 0), 31);
      int iz = min(max((int)floorf((z + 0.5f) * 32.0f), 0), 31);
      voxs[m] = ix * 1024 + iy * 32 + iz;
    }
  }
  __syncthreads();

  // ---- GEMM1: (64 x 160) @ (160 x 1024), fused bias+ReLU+BN -> f16 LDS ----
  for (int tnt = 0; tnt < 8; ++tnt) {
    const int nt = wave * 8 + tnt;           // this wave's 16-wide channel tile
    v8f acc[MT] = {};
#pragma unroll
    for (int kt = 0; kt < KT1; ++kt) {
      v16h bmat = *reinterpret_cast<const v16h*>(w1p + (((kt * NT1 + nt) << 9) + (lane << 4)));
#pragma unroll
      for (int mt = 0; mt < MT; ++mt) {
        v16h amat = lds_load_a(Xs + mt * 16 * XSTRIDE, XSTRIDE, mrow, hi8, kt * 32);
        acc[mt] = __builtin_amdgcn_wmma_f32_16x16x32_f16(false, amat, false, bmat,
                                                         (short)0, acc[mt], false, false);
      }
    }
    const int c  = nt * 16 + mrow;
    const float b1c  = b1[c];
    const float s1c  = g1[c] * rsqrtf(v1[c] + BN_EPS);
    const float m1c  = m1[c];
    const float be1c = be1[c];
#pragma unroll
    for (int mt = 0; mt < MT; ++mt) {
#pragma unroll
      for (int r = 0; r < 8; ++r) {
        const int mr = mt * 16 + r + hi8;
        float h = fmaxf(acc[mt][r] + b1c, 0.0f);
        h = (h - m1c) * s1c + be1c;
        H1s[mr * HSTRIDE + c] = (_Float16)h;
      }
    }
  }
  __syncthreads();

  // ---- GEMM2: (64 x 1024) @ (1024 x 128), fused bias+ReLU+BN ----
  const int nt2 = wave;                      // one 16-wide output tile per wave
  v8f acc2[MT] = {};
#pragma unroll 4
  for (int kt = 0; kt < KT2; ++kt) {
    v16h bmat = *reinterpret_cast<const v16h*>(w2p + (((kt * NT2 + nt2) << 9) + (lane << 4)));
#pragma unroll
    for (int mt = 0; mt < MT; ++mt) {
      v16h amat = lds_load_a(H1s + mt * 16 * HSTRIDE, HSTRIDE, mrow, hi8, kt * 32);
      acc2[mt] = __builtin_amdgcn_wmma_f32_16x16x32_f16(false, amat, false, bmat,
                                                        (short)0, acc2[mt], false, false);
    }
  }
  const int c2 = nt2 * 16 + mrow;
  const float b2c  = b2[c2];
  const float s2c  = g2[c2] * rsqrtf(v2[c2] + BN_EPS);
  const float m2c  = m2[c2];
  const float be2c = be2[c2];

  // ---- scatter-max into voxel grid via native u32 atomic max ----
  const size_t chanBase = ((size_t)b * C_OUT + c2) * (size_t)NGRID;
#pragma unroll
  for (int mt = 0; mt < MT; ++mt) {
#pragma unroll
    for (int r = 0; r < 8; ++r) {
      const int mr = mt * 16 + r + hi8;
      float h = fmaxf(acc2[mt][r] + b2c, 0.0f);
      h = (h - m2c) * s2c + be2c;
      atomicMax(outU + chanBase + voxs[mr], enc_f32(h));
    }
  }
}

extern "C" void kernel_launch(void* const* d_in, const int* in_sizes, int n_in,
                              void* d_out, int out_size, void* d_ws, size_t ws_size,
                              hipStream_t stream) {
  (void)in_sizes; (void)n_in; (void)ws_size;
  const float* xyz = (const float*)d_in[0];
  const float* pf  = (const float*)d_in[1];
  const float* W1  = (const float*)d_in[2];
  const float* b1  = (const float*)d_in[3];
  const float* g1  = (const float*)d_in[4];
  const float* be1 = (const float*)d_in[5];
  const float* m1  = (const float*)d_in[6];
  const float* v1  = (const float*)d_in[7];
  const float* W2  = (const float*)d_in[8];
  const float* b2  = (const float*)d_in[9];
  const float* g2  = (const float*)d_in[10];
  const float* be2 = (const float*)d_in[11];
  const float* m2  = (const float*)d_in[12];
  const float* v2  = (const float*)d_in[13];

  _Float16* w1p = (_Float16*)d_ws;                 // 160*1024 halfs = 320 KB
  _Float16* w2p = w1p + KT1 * NT1 * 512;           // 1024*128 halfs = 256 KB
  unsigned* outU = (unsigned*)d_out;

  pack_w1_kernel<<<(KT1 * NT1 * 512 + 255) / 256, 256, 0, stream>>>(W1, w1p);
  pack_w2_kernel<<<(KT2 * NT2 * 512 + 255) / 256, 256, 0, stream>>>(W2, w2p);
  init_out_kernel<<<(out_size + 255) / 256, 256, 0, stream>>>(outU, out_size);

  fused_mlp_scatter_kernel<<<BATCH * (NPTS / MTILE), 256, 0, stream>>>(
      xyz, pf, b1, g1, be1, m1, v1, b2, g2, be2, m2, v2, w1p, w2p, outU);

  finalize_out_kernel<<<(out_size + 255) / 256, 256, 0, stream>>>(outU, (float*)d_out, out_size);
}
